// AutoEncoder_37976100831248
// MI455X (gfx1250) — matrix operand order repaired and checked by reading, not compiled, not run on hardware
//
#include <hip/hip_runtime.h>

typedef __attribute__((ext_vector_type(16))) __bf16 v16bf;
typedef __attribute__((ext_vector_type(8)))  float  v8f;

struct __align__(16) U4 { unsigned int x, y, z, w; };
union BFrag { v16bf v; U4 u[2]; };

#define BATCH 16384
#define MDIM  4096
#define DDIM  784
#define DPAD  800
#define KSEL  64

// ---------------------------------------------------------------------------
// gfx1250 async global->LDS copy (ASYNCcnt-tracked, no VGPR round trip)
//   lds_off: byte offset within the workgroup LDS allocation (low 32 bits of
//            the flat address of a __shared__ object, per ISA 10.2 aperture map)
// ---------------------------------------------------------------------------
__device__ __forceinline__ void async_copy_b128(unsigned lds_off, const void* gptr) {
    asm volatile("global_load_async_to_lds_b128 %0, %1, off"
                 :: "v"(lds_off), "v"(gptr)
                 : "memory");
}
__device__ __forceinline__ void wait_async0() {
    asm volatile("s_wait_asynccnt 0x0" ::: "memory");
}
__device__ __forceinline__ unsigned lds_offset_of(const void* p) {
    return (unsigned)(uintptr_t)p;  // low 32 bits of flat LDS address = LDS offset
}

// ---------------------------------------------------------------------------
// Conversion kernels: stage bf16 copies of x (padded K) and W (two layouts)
// ---------------------------------------------------------------------------
__global__ __launch_bounds__(256) void cvt_x_kernel(const float* __restrict__ x,
                                                    __bf16* __restrict__ xb) {
    size_t idx = (size_t)blockIdx.x * 256 + threadIdx.x;
    if (idx >= (size_t)BATCH * DPAD) return;
    int d = (int)(idx % DPAD);
    size_t r = idx / DPAD;
    float v = (d < DDIM) ? x[r * DDIM + d] : 0.0f;
    xb[idx] = (__bf16)v;
}

__global__ __launch_bounds__(256) void cvt_w_kernel(const float* __restrict__ W,
                                                    __bf16* __restrict__ Wt,
                                                    __bf16* __restrict__ Wd) {
    size_t idx = (size_t)blockIdx.x * 256 + threadIdx.x;
    if (idx >= (size_t)MDIM * DPAD) return;
    int d = (int)(idx % DPAD);
    int m = (int)(idx / DPAD);
    float v = (d < DDIM) ? W[(size_t)d * MDIM + m] : 0.0f;
    Wt[idx] = (__bf16)v;                                  // Wt[m][d]  (M x DPAD)
    if (d < DDIM) Wd[(size_t)d * MDIM + m] = (__bf16)v;   // Wd[d][m]  (DDIM x MDIM)
}

// ---------------------------------------------------------------------------
// GEMM: C[row, n] = sum_k A[row,k] * B[n,k] + bias[n]
//   A: row-major [rows x K] (bf16; f32 converted on load when AF32)
//   Bmat: row-major [N x K] bf16 (B transposed so K is contiguous)
//   8 waves/WG stacked in M; wave tile (MT*16) x (NT*16).
//   B K-panel double-buffered in LDS via async global->LDS copies.
// ---------------------------------------------------------------------------
template <int MT, int NT, bool AF32>
__global__ __launch_bounds__(256)
void gemm_bf16_wmma(const void* __restrict__ Aptr,
                    const __bf16* __restrict__ Bmat,
                    const float* __restrict__ bias,
                    float* __restrict__ C,
                    int lda, int ldb, int ldc, int ksteps) {
    constexpr int LDSROW = 40;            // 32 data + 8 pad bf16 (80B rows)
    constexpr int TILE   = NT * 16 * LDSROW;  // bf16 elems per buffer
    __shared__ __align__(16) __bf16 ldsB[2 * TILE];

    const int tid  = threadIdx.x;
    const int wave = tid >> 5;
    const int lane = tid & 31;
    const int l16  = lane & 15;
    const int half = lane >> 4;

    const int rowBase = blockIdx.x * (8 * MT * 16) + wave * (MT * 16);
    const int nBase   = blockIdx.y * (NT * 16);
    const unsigned ldsBase = lds_offset_of(&ldsB[0]);

    v8f acc[MT][NT] = {};

    // Issue async copies of B panel (NT*16 rows x 32 K) for k-step kt into buf
    auto issue_tile = [&](int kt, int buf) {
        const int k0 = kt * 32;
        for (int c = tid; c < NT * 64; c += 256) {
            const int r  = c >> 2;
            const int ci = c & 3;
            const __bf16* g = Bmat + (size_t)(nBase + r) * ldb + k0 + ci * 8;
            const unsigned lds = ldsBase + (unsigned)(buf * TILE + r * LDSROW + ci * 8) * 2u;
            async_copy_b128(lds, g);
        }
    };

    issue_tile(0, 0);
    wait_async0();
    __syncthreads();

    for (int kt = 0; kt < ksteps; ++kt) {
        const int buf = kt & 1;
        const int k0  = kt * 32;

        if (kt + 1 < ksteps) issue_tile(kt + 1, buf ^ 1);

        // A fragments: 16x32, lane l16 = row M, half selects K ranges
        //   elems 0..7  -> K = k0 + half*8 + e ; elems 8..15 -> +16
        BFrag a[MT];
#pragma unroll
        for (int mi = 0; mi < MT; ++mi) {
            const int arow = rowBase + mi * 16 + l16;
            if constexpr (!AF32) {
                const __bf16* ap = (const __bf16*)Aptr + (size_t)arow * lda + k0 + half * 8;
                a[mi].u[0] = *(const U4*)(ap);
                a[mi].u[1] = *(const U4*)(ap + 16);
            } else {
                const float* ap = (const float*)Aptr + (size_t)arow * lda + k0;
#pragma unroll
                for (int e = 0; e < 8; ++e) a[mi].v[e] = (__bf16)ap[half * 8 + e];
#pragma unroll
                for (int e = 0; e < 8; ++e) a[mi].v[8 + e] = (__bf16)ap[16 + half * 8 + e];
            }
        }

#pragma unroll
        for (int j = 0; j < NT; ++j) {
            // B fragment: 32x16, lane l16 = column N, elems 0..7 -> K = k0 + half*8 + e ; elems 8..15 -> +16
            BFrag b;
            const __bf16* bp = &ldsB[buf * TILE + (j * 16 + l16) * LDSROW + half * 8];
            b.u[0] = *(const U4*)(bp);
            b.u[1] = *(const U4*)(bp + 16);
#pragma unroll
            for (int mi = 0; mi < MT; ++mi) {
                acc[mi][j] = __builtin_amdgcn_wmma_f32_16x16x32_bf16(
                    false, a[mi].v, false, b.v, (short)0, acc[mi][j], false, false);
            }
        }

        if (kt + 1 < ksteps) wait_async0();
        __syncthreads();
    }

    // C/D layout: VGPR v holds M = v + 8*half, N = nBase + j*16 + l16
#pragma unroll
    for (int j = 0; j < NT; ++j) {
        const int n = nBase + j * 16 + l16;
        const float bv = bias[n];
#pragma unroll
        for (int mi = 0; mi < MT; ++mi) {
#pragma unroll
            for (int v = 0; v < 8; ++v) {
                C[(size_t)(rowBase + mi * 16 + half * 8 + v) * ldc + n] = acc[mi][j][v] + bv;
            }
        }
    }
}

// ---------------------------------------------------------------------------
// Per-row exact radix select of the (K+1)-th largest |value|, mask, count nnz
// ---------------------------------------------------------------------------
__global__ __launch_bounds__(256)
void topk_mask_kernel(const float* __restrict__ enc,
                      float* __restrict__ res,
                      unsigned int* __restrict__ nnz) {
    __shared__ unsigned int hist[256];
    __shared__ unsigned int s_sel;
    __shared__ unsigned int s_rem;
    __shared__ unsigned int s_cnt;

    const int row = blockIdx.x;
    const int tid = threadIdx.x;
    const float* erow = enc + (size_t)row * MDIM;

    unsigned int keys[16];
#pragma unroll
    for (int i = 0; i < 16; ++i)
        keys[i] = __float_as_uint(erow[tid + i * 256]) & 0x7FFFFFFFu;

    if (tid == 0) s_rem = KSEL + 1;  // rank 65 from the top
    unsigned int prefix = 0;

    for (int pass = 0; pass < 4; ++pass) {
        const int shift = 24 - pass * 8;
        const unsigned int maskHi = pass ? (0xFFFFFFFFu << (shift + 8)) : 0u;
        hist[tid] = 0;
        __syncthreads();
#pragma unroll
        for (int i = 0; i < 16; ++i)
            if ((keys[i] & maskHi) == prefix)
                atomicAdd(&hist[(keys[i] >> shift) & 255u], 1u);
        __syncthreads();
        if (tid == 0) {
            unsigned int rem = s_rem;
            unsigned int sel = 0;
            for (int b = 255; b >= 0; --b) {
                unsigned int c = hist[b];
                if (rem <= c) { sel = (unsigned int)b; break; }
                rem -= c;
            }
            s_sel = sel;
            s_rem = rem;
        }
        __syncthreads();
        prefix |= (s_sel << shift);
        __syncthreads();  // protect s_sel/s_rem before next pass / final reuse
    }

    const unsigned int T = prefix;  // exact bits of the 65th-largest |value|
    unsigned int cnt = 0;
#pragma unroll
    for (int i = 0; i < 16; ++i) {
        const int idx = tid + i * 256;
        const bool keep = keys[i] > T;  // strictly greater: matches reference
        res[(size_t)row * MDIM + idx] = keep ? erow[idx] : 0.0f;
        cnt += keep ? 1u : 0u;
    }
    if (tid == 0) s_cnt = 0;
    __syncthreads();
    atomicAdd(&s_cnt, cnt);
    __syncthreads();
    if (tid == 0) atomicAdd(nnz, s_cnt);
}

__global__ void nnz_finalize_kernel(const unsigned int* __restrict__ cnt,
                                    float* __restrict__ out) {
    out[0] = (float)(*cnt) / (float)BATCH;
}

// ---------------------------------------------------------------------------
extern "C" void kernel_launch(void* const* d_in, const int* in_sizes, int n_in,
                              void* d_out, int out_size, void* d_ws, size_t ws_size,
                              hipStream_t stream) {
    (void)in_sizes; (void)n_in; (void)out_size; (void)ws_size;
    const float* x  = (const float*)d_in[0];
    const float* W  = (const float*)d_in[1];
    const float* b1 = (const float*)d_in[2];
    const float* b2 = (const float*)d_in[3];
    // d_in[4] is K (=64), hardcoded as KSEL

    float* enc = (float*)d_out;                       // BATCH*MDIM
    float* dec = enc + (size_t)BATCH * MDIM;          // BATCH*DDIM
    float* nnz = dec + (size_t)BATCH * DDIM;          // 1
    float* res = nnz + 1;                             // BATCH*MDIM

    __bf16* xb = (__bf16*)d_ws;                       // BATCH*DPAD
    __bf16* Wt = xb + (size_t)BATCH * DPAD;           // MDIM*DPAD (W^T padded)
    __bf16* Wd = Wt + (size_t)MDIM * DPAD;            // DDIM*MDIM (W original)
    unsigned int* cnt = (unsigned int*)(Wd + (size_t)DDIM * MDIM);

    hipMemsetAsync(cnt, 0, sizeof(unsigned int), stream);

    {
        size_t n = (size_t)BATCH * DPAD;
        cvt_x_kernel<<<dim3((unsigned)((n + 255) / 256)), dim3(256), 0, stream>>>(x, xb);
    }
    {
        size_t n = (size_t)MDIM * DPAD;
        cvt_w_kernel<<<dim3((unsigned)((n + 255) / 256)), dim3(256), 0, stream>>>(W, Wt, Wd);
    }

    // encoded = x @ W + b1   (A = xb [BATCH x DPAD], B = Wt [MDIM x DPAD])
    gemm_bf16_wmma<2, 8, false><<<dim3(BATCH / 256, MDIM / 128), dim3(256), 0, stream>>>(
        xb, Wt, b1, enc, DPAD, DPAD, MDIM, DPAD / 32);

    // top-K mask -> res, nnz count
    topk_mask_kernel<<<dim3(BATCH), dim3(256), 0, stream>>>(enc, res, cnt);

    // decoded = res @ W^T + b2  (A = res f32 [BATCH x MDIM], B = Wd [DDIM x MDIM])
    gemm_bf16_wmma<2, 7, true><<<dim3(BATCH / 256, DDIM / 112), dim3(256), 0, stream>>>(
        res, Wd, b2, dec, MDIM, MDIM, DDIM, MDIM / 32);

    nnz_finalize_kernel<<<1, 1, 0, stream>>>(cnt, nnz);
}
